// GNN_v3_5927054868945
// MI455X (gfx1250) — matrix operand order repaired and checked by reading, not compiled, not run on hardware
//
#include <hip/hip_runtime.h>
#include <math.h>

// ---------------------------------------------------------------------------
// GNN forward (3x GATConv + GraphNorm, 2x ARMAConv + GraphNorm, sort-pool, lin)
// MI455X / gfx1250: dense transforms via LDS-tiled v_wmma_f32_16x16x32_f16
// (f32 accum, 64x64 C tile per 4-wave block, 4 WMMA/wave/k-step, double-
// buffered LDS, weights pre-packed to f16 fragment layout once per layer),
// sparse segment ops via global float atomics, wave32 everywhere.
// ---------------------------------------------------------------------------

typedef __attribute__((ext_vector_type(16))) _Float16 v16h;
typedef __attribute__((ext_vector_type(8)))  _Float16 v8h;
typedef __attribute__((ext_vector_type(8)))  float    v8f;

#define kN    16384      // nodes
#define kE    131072     // edges
#define kESL  (kE + kN)  // edges + self loops (GAT)
#define kNB   64         // graphs
#define kNG   256        // nodes per graph
#define kEPS  1e-5f

#define TM 64            // C macro-tile rows   (4 waves x 16)
#define TN 64            // C macro-tile cols   (4 x 16 per wave)
#define TK 32            // k-step (WMMA K)

// =================== weight pre-pack: f32 [K,N] -> f16 blocked =============
// Bp layout: [n_blk][k_step][n in 0..63][k in 0..31], so each (n_blk,k_step)
// tile is a contiguous 4KB block matching the GEMM's LDS B layout exactly.
__global__ void k_pack_B(const float* __restrict__ Bw, _Float16* __restrict__ Bp,
                         int K, int Nn) {
  long i = blockIdx.x * (long)blockDim.x + threadIdx.x;
  if (i >= (long)K * Nn) return;
  int k = (int)(i / Nn), n = (int)(i % Nn);
  int nb = n / TN, nn = n % TN, ks = k / TK, kk = k % TK;
  Bp[((size_t)nb * (K / TK) + ks) * (TN * TK) + nn * TK + kk] = (_Float16)Bw[i];
}

// ============================ WMMA GEMM ====================================
// C[M,N] = A[M,K] @ B[K,N]; A fp32, B pre-packed f16, f32 accumulation.
// Requires M%64==0, N%64==0, K%32==0 (true for all layers).
__global__ void __launch_bounds__(128)
k_wmma_gemm(const float* __restrict__ A, const _Float16* __restrict__ Bp,
            float* __restrict__ C, int M, int K, int Nn) {
  __shared__ __align__(16) _Float16 lds_A[2][TM * TK];  // [m][k]
  __shared__ __align__(16) _Float16 lds_B[2][TN * TK];  // [n][k]

  const int tid  = threadIdx.x;
  const int lane = tid & 31;
  const int wave = tid >> 5;
  const int m0 = blockIdx.x * TM;
  const int n0 = blockIdx.y * TN;
  const int col  = lane & 15;
  const int half = lane >> 4;
  const int nsteps = K / TK;
  const _Float16* Bblk = Bp + (size_t)(n0 / TN) * nsteps * (TN * TK);

  v8f acc[4] = {v8f{}, v8f{}, v8f{}, v8f{}};

  // ---- prologue: stage tile 0 into buffer 0 ----
  {
#pragma unroll
    for (int j = 0; j < 4; ++j) {
      int idx = tid + 128 * j;         // float4 index, 512 total
      int r   = idx >> 3;              // row 0..63
      int c4  = idx & 7;               // float4-col 0..7
      const float4 v = *(const float4*)(A + (size_t)(m0 + r) * K + c4 * 4);
      _Float16* d = &lds_A[0][r * TK + c4 * 4];
      d[0] = (_Float16)v.x; d[1] = (_Float16)v.y;
      d[2] = (_Float16)v.z; d[3] = (_Float16)v.w;
    }
    *(v16h*)&lds_B[0][tid * 16] = *(const v16h*)&Bblk[tid * 16];
  }
  __syncthreads();

  for (int s = 0; s < nsteps; ++s) {
    const int buf  = s & 1;
    const bool more = (s + 1) < nsteps;

    // ---- issue next tile's global loads first (latency under the WMMAs) ----
    float4 va[4];
    v16h   vb;
    if (more) {
#pragma unroll
      for (int j = 0; j < 4; ++j) {
        int idx = tid + 128 * j;
        int r   = idx >> 3;
        int c4  = idx & 7;
        va[j] = *(const float4*)(A + (size_t)(m0 + r) * K + (s + 1) * TK + c4 * 4);
      }
      vb = *(const v16h*)&Bblk[(size_t)(s + 1) * (TN * TK) + tid * 16];
    }
    if (s + 2 < nsteps)
      __builtin_prefetch(A + (size_t)(m0 + (tid >> 1)) * K + (s + 2) * TK, 0, 3);

    // ---- A fragment (16x32): one contiguous 16-half run per lane (k = 16*half + s) ----
    const int m = wave * 16 + col;
    v8h alo = *(const v8h*)&lds_A[buf][m * TK + 16 * half];
    v8h ahi = *(const v8h*)&lds_A[buf][m * TK + 16 * half + 8];
    v16h afrag = __builtin_shufflevector(alo, ahi, 0, 1, 2, 3, 4, 5, 6, 7,
                                         8, 9, 10, 11, 12, 13, 14, 15);
    // ---- 4 WMMAs: B fragment is one contiguous 16-half run per lane ----
#pragma unroll
    for (int nt = 0; nt < 4; ++nt) {
      v16h bfrag = *(const v16h*)&lds_B[buf][(nt * 16 + col) * TK + 16 * half];
      acc[nt] = __builtin_amdgcn_wmma_f32_16x16x32_f16(
          false, afrag, false, bfrag, (short)0, acc[nt], false, false);
    }

    // ---- convert + store next tile into the other buffer ----
    if (more) {
#pragma unroll
      for (int j = 0; j < 4; ++j) {
        int idx = tid + 128 * j;
        int r   = idx >> 3;
        int c4  = idx & 7;
        _Float16* d = &lds_A[buf ^ 1][r * TK + c4 * 4];
        d[0] = (_Float16)va[j].x; d[1] = (_Float16)va[j].y;
        d[2] = (_Float16)va[j].z; d[3] = (_Float16)va[j].w;
      }
      *(v16h*)&lds_B[buf ^ 1][tid * 16] = vb;
    }
    __syncthreads();
  }

  // ---- store: C/D layout VGPR r -> M = r + 8*half, N = col ----
#pragma unroll
  for (int nt = 0; nt < 4; ++nt) {
#pragma unroll
    for (int r = 0; r < 8; ++r)
      C[(size_t)(m0 + wave * 16 + r + 8 * half) * Nn + (n0 + nt * 16 + col)] =
          acc[nt][r];
  }
}

// ============================ small helpers ================================
__global__ void k_fill_f(float* p, float v, long n) {
  long i = blockIdx.x * (long)blockDim.x + threadIdx.x;
  if (i < n) p[i] = v;
}
__global__ void k_fill_u(unsigned* p, unsigned v, long n) {
  long i = blockIdx.x * (long)blockDim.x + threadIdx.x;
  if (i < n) p[i] = v;
}

// h = x * W  for GAT1 (fan_in == 1):  h[n][f] = x[n] * W[0][f]
__global__ void k_h1(const float* __restrict__ x, const float* __restrict__ W,
                     float* __restrict__ h) {
  long i = blockIdx.x * (long)blockDim.x + threadIdx.x;
  if (i < (long)kN * 64) {
    int f = (int)(i & 63);
    int n = (int)(i >> 6);
    h[i] = x[n] * W[f];
  }
}

// degree / gcn_norm (no self loops, ARMA path)
__global__ void k_deg(const int* __restrict__ dst, float* __restrict__ deg) {
  int e = blockIdx.x * blockDim.x + threadIdx.x;
  if (e < kE) atomicAdd(&deg[dst[e]], 1.0f);
}
__global__ void k_dis(const float* __restrict__ deg, float* __restrict__ dis) {
  int n = blockIdx.x * blockDim.x + threadIdx.x;
  if (n < kN) dis[n] = deg[n] > 0.0f ? rsqrtf(fmaxf(deg[n], 1.0f)) : 0.0f;
}
__global__ void k_enorm(const int* __restrict__ src, const int* __restrict__ dst,
                        const float* __restrict__ dis, float* __restrict__ en) {
  int e = blockIdx.x * blockDim.x + threadIdx.x;
  if (e < kE) en[e] = dis[src[e]] * dis[dst[e]];
}

// ============================ GAT edge machinery ===========================
// per-node attention scores: s_src = h.a_s, s_dst = h.a_d
__global__ void k_scores(const float* __restrict__ h, const float* __restrict__ as_,
                         const float* __restrict__ ad_, float* __restrict__ s_src,
                         float* __restrict__ s_dst, int F) {
  __shared__ float r1[256], r2[256];
  int n = blockIdx.x;
  float a1 = 0.f, a2 = 0.f;
  for (int f = threadIdx.x; f < F; f += blockDim.x) {
    float v = h[(size_t)n * F + f];
    a1 += v * as_[f];
    a2 += v * ad_[f];
  }
  r1[threadIdx.x] = a1; r2[threadIdx.x] = a2;
  __syncthreads();
  for (int s = blockDim.x >> 1; s > 0; s >>= 1) {
    if (threadIdx.x < (unsigned)s) {
      r1[threadIdx.x] += r1[threadIdx.x + s];
      r2[threadIdx.x] += r2[threadIdx.x + s];
    }
    __syncthreads();
  }
  if (threadIdx.x == 0) { s_src[n] = r1[0]; s_dst[n] = r2[0]; }
}

__device__ __forceinline__ unsigned ord_enc(float f) {
  unsigned u = __float_as_uint(f);
  return (u & 0x80000000u) ? ~u : (u | 0x80000000u);
}
__device__ __forceinline__ float ord_dec(unsigned u) {
  u = (u & 0x80000000u) ? (u & 0x7FFFFFFFu) : ~u;
  return __uint_as_float(u);
}

// alpha = leaky_relu(s_src[s] + s_dst[d], 0.2); segment max over dst
__global__ void k_alpha_max(const int* __restrict__ src, const int* __restrict__ dst,
                            const float* __restrict__ s_src, const float* __restrict__ s_dst,
                            float* __restrict__ alpha, unsigned* __restrict__ amax) {
  int e = blockIdx.x * blockDim.x + threadIdx.x;
  if (e >= kESL) return;
  int s = (e < kE) ? src[e] : (e - kE);
  int d = (e < kE) ? dst[e] : (e - kE);
  float a = s_src[s] + s_dst[d];
  a = (a > 0.f) ? a : 0.2f * a;
  alpha[e] = a;
  atomicMax(&amax[d], ord_enc(a));
}

// ex = exp(alpha - amax[dst]); segment sum over dst
__global__ void k_exp_sum(const int* __restrict__ src, const int* __restrict__ dst,
                          const float* __restrict__ alpha, const unsigned* __restrict__ amax,
                          float* __restrict__ exb, float* __restrict__ denom) {
  int e = blockIdx.x * blockDim.x + threadIdx.x;
  if (e >= kESL) return;
  int d = (e < kE) ? dst[e] : (e - kE);
  float ex = expf(alpha[e] - ord_dec(amax[d]));
  exb[e] = ex;
  atomicAdd(&denom[d], ex);
}

// out init: out[n][f] = b[f]  (GAT adds bias after aggregation)
__global__ void k_bias_init(float* __restrict__ out, const float* __restrict__ b,
                            long tot, int F) {
  long i = blockIdx.x * (long)blockDim.x + threadIdx.x;
  if (i < tot) out[i] = b[(int)(i % F)];
}

// out[dst] += h[src] * (ex/denom[dst]); one block per edge, threads over feats
__global__ void k_gat_agg(const int* __restrict__ src, const int* __restrict__ dst,
                          const float* __restrict__ h, const float* __restrict__ exb,
                          const float* __restrict__ denom, float* __restrict__ out, int F) {
  int e = blockIdx.x;
  int f = blockIdx.y * blockDim.x + threadIdx.x;
  int s = (e < kE) ? src[e] : (e - kE);
  int d = (e < kE) ? dst[e] : (e - kE);
  float w = exb[e] / denom[d];
  atomicAdd(&out[(size_t)d * F + f], h[(size_t)s * F + f] * w);
}

__global__ void k_elu(float* __restrict__ x, long tot) {
  long i = blockIdx.x * (long)blockDim.x + threadIdx.x;
  if (i < tot) {
    float v = x[i];
    x[i] = v > 0.f ? v : (expf(v) - 1.f);
  }
}

// ============================ ARMA edge machinery ==========================
__global__ void k_arma_agg(const int* __restrict__ src, const int* __restrict__ dst,
                           const float* __restrict__ h, const float* __restrict__ en,
                           float* __restrict__ agg, int F) {
  int e = blockIdx.x;
  int f = blockIdx.y * blockDim.x + threadIdx.x;
  atomicAdd(&agg[(size_t)dst[e] * F + f], h[(size_t)src[e] * F + f] * en[e]);
}
__global__ void k_relu_bias(const float* __restrict__ agg, const float* __restrict__ b,
                            float* __restrict__ out, long tot, int F) {
  long i = blockIdx.x * (long)blockDim.x + threadIdx.x;
  if (i < tot) out[i] = fmaxf(agg[i] + b[(int)(i % F)], 0.f);
}

// ============================ GraphNorm ====================================
__global__ void k_gn_mean(const float* __restrict__ x, float* __restrict__ mean, int F) {
  int g = blockIdx.x;
  int f = blockIdx.y * blockDim.x + threadIdx.x;
  float acc = 0.f;
  for (int n = 0; n < kNG; ++n) acc += x[((size_t)g * kNG + n) * F + f];
  mean[(size_t)g * F + f] = acc * (1.0f / kNG);
}
__global__ void k_gn_center(float* __restrict__ x, const float* __restrict__ mean,
                            const float* __restrict__ ms, int F, long tot) {
  long i = blockIdx.x * (long)blockDim.x + threadIdx.x;
  if (i < tot) {
    int f = (int)(i % F);
    int g = (int)((i / F) >> 8);
    x[i] -= mean[(size_t)g * F + f] * ms[f];
  }
}
__global__ void k_gn_var(const float* __restrict__ x, float* __restrict__ var, int F) {
  int g = blockIdx.x;
  int f = blockIdx.y * blockDim.x + threadIdx.x;
  float acc = 0.f;
  for (int n = 0; n < kNG; ++n) {
    float v = x[((size_t)g * kNG + n) * F + f];
    acc += v * v;
  }
  var[(size_t)g * F + f] = acc * (1.0f / kNG);
}
__global__ void k_gn_scale(float* __restrict__ x, const float* __restrict__ var,
                           const float* __restrict__ w, const float* __restrict__ b,
                           int F, long tot) {
  long i = blockIdx.x * (long)blockDim.x + threadIdx.x;
  if (i < tot) {
    int f = (int)(i % F);
    int g = (int)((i / F) >> 8);
    x[i] = w[f] * x[i] * rsqrtf(var[(size_t)g * F + f] + kEPS) + b[f];
  }
}

// ======================= pooling + sort + final linear =====================
// one block (64 threads) per graph; x is [N,64]
__global__ void k_pool(const float* __restrict__ x, const float* __restrict__ linW,
                       const float* __restrict__ linb, float* __restrict__ out) {
  __shared__ float feat[512];
  __shared__ int   topi[5];
  int g = blockIdx.x;
  int t = threadIdx.x;  // feature 0..63
  float mx = -INFINITY, sm = 0.f;
  for (int n = 0; n < kNG; ++n) {
    float v = x[((size_t)g * kNG + n) * 64 + t];
    mx = fmaxf(mx, v);
    sm += v;
  }
  feat[t]       = mx;                // x_max
  feat[64 + t]  = sm * (1.f / kNG);  // x_mean
  feat[128 + t] = sm;                // x_add
  __syncthreads();
  if (t == 0) {  // top-5 of last channel, desc, ties -> lower index (lax.top_k)
    for (int k = 0; k < 5; ++k) {
      float best = -INFINITY; int bi = 0;
      for (int n = 0; n < kNG; ++n) {
        bool skip = false;
        for (int j = 0; j < k; ++j) if (topi[j] == n) skip = true;
        float v = x[((size_t)g * kNG + n) * 64 + 63];
        if (!skip && v > best) { best = v; bi = n; }
      }
      topi[k] = bi;
    }
  }
  __syncthreads();
  for (int k = 0; k < 5; ++k)
    feat[192 + k * 64 + t] = x[((size_t)g * kNG + topi[k]) * 64 + t];
  __syncthreads();
  if (t < 2) {
    float acc = linb[t];
    for (int i = 0; i < 512; ++i) acc += feat[i] * linW[i * 2 + t];
    out[g * 2 + t] = acc;
  }
}

// ============================ host orchestration ===========================
struct Scratch {
  float *P0, *P1;
  float *s_src, *s_dst, *denom, *alpha, *exb, *deg, *dis, *enorm, *meanb, *varb;
  unsigned *amax;
  _Float16 *Bp;
};

static inline int ceil_div_l(long a, int b) { return (int)((a + b - 1) / b); }

static void gemm(const float* A, const float* Bw, float* C, int M, int K, int Nn,
                 const Scratch& S, hipStream_t s) {
  long tot = (long)K * Nn;
  k_pack_B<<<ceil_div_l(tot, 256), 256, 0, s>>>(Bw, S.Bp, K, Nn);
  k_wmma_gemm<<<dim3(M / TM, Nn / TN), 128, 0, s>>>(A, S.Bp, C, M, K, Nn);
}

static void run_gat_edges(const float* h, int F, const float* as_, const float* ad_,
                          const float* b, const int* srcI, const int* dstI,
                          const Scratch& S, float* outbuf, hipStream_t s) {
  k_scores<<<kN, 256, 0, s>>>(h, as_, ad_, S.s_src, S.s_dst, F);
  k_fill_u<<<ceil_div_l(kN, 256), 256, 0, s>>>(S.amax, 0u, kN);
  k_fill_f<<<ceil_div_l(kN, 256), 256, 0, s>>>(S.denom, 0.f, kN);
  k_alpha_max<<<ceil_div_l(kESL, 256), 256, 0, s>>>(srcI, dstI, S.s_src, S.s_dst,
                                                    S.alpha, S.amax);
  k_exp_sum<<<ceil_div_l(kESL, 256), 256, 0, s>>>(srcI, dstI, S.alpha, S.amax,
                                                  S.exb, S.denom);
  long tot = (long)kN * F;
  k_bias_init<<<ceil_div_l(tot, 256), 256, 0, s>>>(outbuf, b, tot, F);
  int bs = F < 256 ? F : 256;
  k_gat_agg<<<dim3(kESL, F / bs), bs, 0, s>>>(srcI, dstI, h, S.exb, S.denom, outbuf, F);
  k_elu<<<ceil_div_l(tot, 256), 256, 0, s>>>(outbuf, tot);
}

static void run_gnorm(float* xbuf, int F, const float* w, const float* b,
                      const float* ms, const Scratch& S, hipStream_t s) {
  int bs = F < 256 ? F : 256;
  dim3 gr(kNB, F / bs);
  long tot = (long)kN * F;
  int eb = ceil_div_l(tot, 256);
  k_gn_mean<<<gr, bs, 0, s>>>(xbuf, S.meanb, F);
  k_gn_center<<<eb, 256, 0, s>>>(xbuf, S.meanb, ms, F, tot);
  k_gn_var<<<gr, bs, 0, s>>>(xbuf, S.varb, F);
  k_gn_scale<<<eb, 256, 0, s>>>(xbuf, S.varb, w, b, F, tot);
}

extern "C" void kernel_launch(void* const* d_in, const int* in_sizes, int n_in,
                              void* d_out, int out_size, void* d_ws, size_t ws_size,
                              hipStream_t stream) {
  (void)in_sizes; (void)n_in; (void)out_size; (void)ws_size;
  // ---- inputs, in setup_inputs() dict (insertion) order ----
  const float* x        = (const float*)d_in[0];
  const float* gat_W[3]  = {(const float*)d_in[1], (const float*)d_in[5], (const float*)d_in[9]};
  const float* gat_as[3] = {(const float*)d_in[2], (const float*)d_in[6], (const float*)d_in[10]};
  const float* gat_ad[3] = {(const float*)d_in[3], (const float*)d_in[7], (const float*)d_in[11]};
  const float* gat_b[3]  = {(const float*)d_in[4], (const float*)d_in[8], (const float*)d_in[12]};
  const float* gn_w[5]  = {(const float*)d_in[13], (const float*)d_in[16], (const float*)d_in[19],
                           (const float*)d_in[22], (const float*)d_in[25]};
  const float* gn_b[5]  = {(const float*)d_in[14], (const float*)d_in[17], (const float*)d_in[20],
                           (const float*)d_in[23], (const float*)d_in[26]};
  const float* gn_ms[5] = {(const float*)d_in[15], (const float*)d_in[18], (const float*)d_in[21],
                           (const float*)d_in[24], (const float*)d_in[27]};
  const float* arma1_Wi = (const float*)d_in[28];
  const float* arma1_Wr = (const float*)d_in[29];
  const float* arma1_b  = (const float*)d_in[30];
  const float* arma2_Wi = (const float*)d_in[31];
  const float* arma2_Wr = (const float*)d_in[32];
  const float* arma2_b  = (const float*)d_in[33];
  const float* lin_W    = (const float*)d_in[34];
  const float* lin_b    = (const float*)d_in[35];
  const int*   ei       = (const int*)d_in[36];
  const int*   srcI = ei;
  const int*   dstI = ei + kE;
  float* out = (float*)d_out;

  // ---- workspace carve ----
  char* w = (char*)d_ws;
  Scratch S;
  S.P0    = (float*)w; w += (size_t)kN * 2048 * 4;
  S.P1    = (float*)w; w += (size_t)kN * 2048 * 4;
  S.s_src = (float*)w; w += (size_t)kN * 4;
  S.s_dst = (float*)w; w += (size_t)kN * 4;
  S.amax  = (unsigned*)w; w += (size_t)kN * 4;
  S.denom = (float*)w; w += (size_t)kN * 4;
  S.alpha = (float*)w; w += (size_t)kESL * 4;
  S.exb   = (float*)w; w += (size_t)kESL * 4;
  S.deg   = (float*)w; w += (size_t)kN * 4;
  S.dis   = (float*)w; w += (size_t)kN * 4;
  S.enorm = (float*)w; w += (size_t)kE * 4;
  S.meanb = (float*)w; w += (size_t)kNB * 2048 * 4;
  S.varb  = (float*)w; w += (size_t)kNB * 2048 * 4;
  S.Bp    = (_Float16*)w; w += (size_t)2048 * 2048 * 2;  // packed weights (8MB: 2048x2048 f16)

  // ---- gcn_norm for ARMA (no self loops) ----
  k_fill_f<<<ceil_div_l(kN, 256), 256, 0, stream>>>(S.deg, 0.f, kN);
  k_deg<<<ceil_div_l(kE, 256), 256, 0, stream>>>(dstI, S.deg);
  k_dis<<<ceil_div_l(kN, 256), 256, 0, stream>>>(S.deg, S.dis);
  k_enorm<<<ceil_div_l(kE, 256), 256, 0, stream>>>(srcI, dstI, S.dis, S.enorm);

  // ---- GAT1: 1 -> 64 ----
  k_h1<<<ceil_div_l((long)kN * 64, 256), 256, 0, stream>>>(x, gat_W[0], S.P0);
  run_gat_edges(S.P0, 64, gat_as[0], gat_ad[0], gat_b[0], srcI, dstI, S, S.P1, stream);
  run_gnorm(S.P1, 64, gn_w[0], gn_b[0], gn_ms[0], S, stream);

  // ---- GAT2: 64 -> 512 ----
  gemm(S.P1, gat_W[1], S.P0, kN, 64, 512, S, stream);
  run_gat_edges(S.P0, 512, gat_as[1], gat_ad[1], gat_b[1], srcI, dstI, S, S.P1, stream);
  run_gnorm(S.P1, 512, gn_w[1], gn_b[1], gn_ms[1], S, stream);

  // ---- GAT3: 512 -> 2048 ----
  gemm(S.P1, gat_W[2], S.P0, kN, 512, 2048, S, stream);
  run_gat_edges(S.P0, 2048, gat_as[2], gat_ad[2], gat_b[2], srcI, dstI, S, S.P1, stream);
  run_gnorm(S.P1, 2048, gn_w[2], gn_b[2], gn_ms[2], S, stream);

  // ---- ARMA1: 2048 -> 512 ----
  {
    float* h1  = S.P0;                        // X @ Wi
    float* agg = S.P0 + (size_t)kN * 512;     // X @ Wr, then += aggregation
    gemm(S.P1, arma1_Wi, h1, kN, 2048, 512, S, stream);
    gemm(S.P1, arma1_Wr, agg, kN, 2048, 512, S, stream);
    k_arma_agg<<<dim3(kE, 2), 256, 0, stream>>>(srcI, dstI, h1, S.enorm, agg, 512);
    long tot = (long)kN * 512;
    k_relu_bias<<<ceil_div_l(tot, 256), 256, 0, stream>>>(agg, arma1_b, S.P1, tot, 512);
    run_gnorm(S.P1, 512, gn_w[3], gn_b[3], gn_ms[3], S, stream);
  }

  // ---- ARMA2: 512 -> 64 ----
  {
    float* h1  = S.P0;
    float* agg = S.P0 + (size_t)kN * 64;
    gemm(S.P1, arma2_Wi, h1, kN, 512, 64, S, stream);
    gemm(S.P1, arma2_Wr, agg, kN, 512, 64, S, stream);
    k_arma_agg<<<dim3(kE, 1), 64, 0, stream>>>(srcI, dstI, h1, S.enorm, agg, 64);
    long tot = (long)kN * 64;
    k_relu_bias<<<ceil_div_l(tot, 256), 256, 0, stream>>>(agg, arma2_b, S.P1, tot, 64);
    run_gnorm(S.P1, 64, gn_w[4], gn_b[4], gn_ms[4], S, stream);
  }

  // ---- pooling (max/mean/add/sort-top5) + final linear 512 -> 2 ----
  k_pool<<<kNB, 64, 0, stream>>>(S.P1, lin_W, lin_b, out);
}
